// GMM_47742856462401
// MI455X (gfx1250) — matrix operand order repaired and checked by reading, not compiled, not run on hardware
//
#include <hip/hip_runtime.h>
#include <hip/hip_bf16.h>
#include <cstdint>

// ---------------------------------------------------------------------------
// GMM log-likelihood:  quad[n,k] = (x^2)·iv + x·(-2 mu iv) + c[k]
// Fused GEMM  A[N,512] x B[512,64]  in split-precision bf16 WMMA
// (a_hi*b_hi + a_hi*b_lo + a_lo*b_hi, f32 accumulate), then logsumexp over
// K=64 per row and a deterministic mean reduction.
//
// v3: (a) conflict-free LDS layout for B fragments — each 32B fragment is
// split into two 16B halves in separate 128-dword planes, so a ds_load_b128
// has lane-stride 4 dwords (32 lanes -> 128 consecutive dwords, no bank
// conflicts; the old layout had a 4-way conflict). (b) force full unroll of
// the d-block loop so the scheduler can software-pipeline LDS loads across
// WMMAs instead of s_wait_dscnt 0x0 stalls per tile.
// ---------------------------------------------------------------------------

typedef __attribute__((ext_vector_type(16))) __bf16 v16bf;
typedef __attribute__((ext_vector_type(8)))  float  v8f;

#define GMM_D 256
#define GMM_K 64
#define GMM_LOG2PI 1.8378770664093453f

// split float into bf16 hi + bf16 lo (RNE conversions)
static __device__ inline void split_bf16(float v, unsigned short& hi, unsigned short& lo) {
    __bf16 h = (__bf16)v;
    float hf = (float)h;
    __bf16 l = (__bf16)(v - hf);
    union { __bf16 b; unsigned short u; } ch, cl;
    ch.b = h; cl.b = l;
    hi = ch.u; lo = cl.u;
}

// B value at (inner index kk in [0,512), component col):
//   kk <  256 : inv_var[col][kk]
//   kk >= 256 : -2 * mu[col][kk-256] * inv_var[col][kk-256]
static __device__ inline float bval(const float* __restrict__ mu,
                                    const float* __restrict__ lc,
                                    int col, int kk) {
    if (kk < GMM_D) {
        return __expf(-lc[col * GMM_D + kk]);
    } else {
        int d = kk - GMM_D;
        float iv = __expf(-lc[col * GMM_D + d]);
        return -2.0f * mu[col * GMM_D + d] * iv;
    }
}

// ---------------------------------------------------------------------------
// Prep 1: pack B into WMMA B-fragment order, split bf16 hi/lo, stored in a
// bank-conflict-free "half plane" layout:
//   dword index = ((kb*4 + c)*2 + h)*128 + lane*4 + j      h,j: halves
// The dword at (h, j) holds v16bf elements e = 2*(h*4+j), e+1 with
//   K = kb*32 + (lane<16 ? 0 : 8) + (e<8 ? e : 8+e)   and column col = c*16 + (lane&15).
// 16 k-blocks * 4 col-tiles * 2 halves * 32 lanes * 4 dwords = 16384 dwords.
// ---------------------------------------------------------------------------
__global__ void __launch_bounds__(256)
gmm_prep_frags(const float* __restrict__ mu, const float* __restrict__ lc,
               uint32_t* __restrict__ bhi, uint32_t* __restrict__ blo) {
    int t = blockIdx.x * 256 + threadIdx.x;      // 0 .. 16383
    int j    = t & 3;
    int lane = (t >> 2) & 31;
    int h    = (t >> 7) & 1;
    int c    = (t >> 8) & 3;
    int kb   = t >> 10;

    int col = c * 16 + (lane & 15);
    int e   = 2 * (h * 4 + j);                   // fragment element index (even)
    int kk  = kb * 32 + ((lane < 16) ? 0 : 8) + ((e < 8) ? e : 8 + e);

    float v0 = bval(mu, lc, col, kk);
    float v1 = bval(mu, lc, col, kk + 1);
    unsigned short h0, l0, h1, l1;
    split_bf16(v0, h0, l0);
    split_bf16(v1, h1, l1);
    bhi[t] = (uint32_t)h0 | ((uint32_t)h1 << 16);
    blo[t] = (uint32_t)l0 | ((uint32_t)l1 << 16);
}

// ---------------------------------------------------------------------------
// Prep 2: per-component constant
//   ck[k] = -0.5*(D*log2pi + sum(lc) + sum(mu^2 * iv)) + log(w + 1e-10)
// ---------------------------------------------------------------------------
__global__ void gmm_prep_const(const float* __restrict__ w,
                               const float* __restrict__ mu,
                               const float* __restrict__ lc,
                               float* __restrict__ ck) {
    int k = threadIdx.x;
    if (k >= GMM_K) return;
    float sld = 0.0f, smi = 0.0f;
    for (int d = 0; d < GMM_D; ++d) {
        float l = lc[k * GMM_D + d];
        float m = mu[k * GMM_D + d];
        sld += l;
        smi += m * m * __expf(-l);
    }
    ck[k] = -0.5f * ((float)GMM_D * GMM_LOG2PI + sld + smi) + logf(w[k] + 1e-10f);
}

// load one v16bf B fragment from the half-plane LDS layout (conflict-free)
static __device__ inline v16bf ld_bfrag(const uint32_t* __restrict__ base,
                                        int kb, int c, int lane) {
    const uint32_t* p0 = base + (size_t)((kb * 4 + c) * 2 + 0) * 128 + lane * 4;
    const uint32_t* p1 = base + (size_t)((kb * 4 + c) * 2 + 1) * 128 + lane * 4;
    union { uint4 q[2]; v16bf v; } u;
    u.q[0] = *(const uint4*)p0;
    u.q[1] = *(const uint4*)p1;
    return u.v;
}

// ---------------------------------------------------------------------------
// Main: 8 waves * 32 rows = 256 rows per workgroup, all 64 components.
// B hi/lo fragments (128 KB) staged in dynamic LDS; each B fragment read
// feeds two 16-row tiles.
// ---------------------------------------------------------------------------
__global__ void __launch_bounds__(256)
gmm_main(const float* __restrict__ x,
         const uint32_t* __restrict__ bhi, const uint32_t* __restrict__ blo,
         const float* __restrict__ ck, float* __restrict__ partials, int N) {
    extern __shared__ uint32_t lds[];     // [0..16383]=B_hi  [16384..32767]=B_lo
    __shared__ float wsum_arr[8];

    const int tid = threadIdx.x;

    // stage B fragments into LDS (coalesced b128 copies)
    for (int i = tid * 4; i < 16384; i += 1024) {
        *(uint4*)&lds[i]         = *(const uint4*)&bhi[i];
        *(uint4*)&lds[16384 + i] = *(const uint4*)&blo[i];
    }
    __syncthreads();

    const int lane = tid & 31;
    const int wave = tid >> 5;
    const int col0 = lane & 15;
    const int base = blockIdx.x * 256 + wave * 32;  // 32 rows per wave
    const int loff = (lane >= 16) ? 8 : 0;

    const int r0 = base + col0;
    const int r1 = base + 16 + col0;
    const int r0c = (r0 < N) ? r0 : (N - 1);        // clamp: WMMA loop stays uniform
    const int r1c = (r1 < N) ? r1 : (N - 1);
    const float* __restrict__ xr0 = x + (size_t)r0c * GMM_D;
    const float* __restrict__ xr1 = x + (size_t)r1c * GMM_D;

    const uint32_t* __restrict__ ldsHi = lds;
    const uint32_t* __restrict__ ldsLo = lds + 16384;

    v8f acc[2][4] = {};

#pragma unroll
    for (int db = 0; db < 8; ++db) {
        const int dbase = db * 32 + loff;

        // Build all four A-fragment pairs for this d-block: squared (k-block
        // db) and plain (k-block db+8), for both 16-row tiles. Each x value
        // is loaded exactly once.
        v16bf sh[2], sl[2], xh[2], xl[2];
#pragma unroll
        for (int r = 0; r < 2; ++r) {
            const float* __restrict__ xr = r ? xr1 : xr0;
            // lane<16 : K offsets {0..7,16..23}; lane>=16: {8..15,24..31}
            float4 p0 = *(const float4*)(xr + dbase);
            float4 p1 = *(const float4*)(xr + dbase + 4);
            float4 p2 = *(const float4*)(xr + dbase + 16);
            float4 p3 = *(const float4*)(xr + dbase + 20);
            float f[16] = { p0.x, p0.y, p0.z, p0.w,  p1.x, p1.y, p1.z, p1.w,
                            p2.x, p2.y, p2.z, p2.w,  p3.x, p3.y, p3.z, p3.w };
#pragma unroll
            for (int e = 0; e < 16; ++e) {
                float vx = f[e];
                __bf16 hx = (__bf16)vx;
                xh[r][e] = hx;
                xl[r][e] = (__bf16)(vx - (float)hx);
                float vs = vx * vx;
                __bf16 hs = (__bf16)vs;
                sh[r][e] = hs;
                sl[r][e] = (__bf16)(vs - (float)hs);
            }
        }

#pragma unroll
        for (int c = 0; c < 4; ++c) {
            v16bf bsh = ld_bfrag(ldsHi, db,     c, lane);   // B for x^2 rows, hi
            v16bf bsl = ld_bfrag(ldsLo, db,     c, lane);   // B for x^2 rows, lo
            v16bf bxh = ld_bfrag(ldsHi, db + 8, c, lane);   // B for x rows,  hi
            v16bf bxl = ld_bfrag(ldsLo, db + 8, c, lane);   // B for x rows,  lo
#pragma unroll
            for (int r = 0; r < 2; ++r) {
                v8f A = acc[r][c];
                A = __builtin_amdgcn_wmma_f32_16x16x32_bf16(false, sh[r], false, bsh, (short)0, A, false, false);
                A = __builtin_amdgcn_wmma_f32_16x16x32_bf16(false, sh[r], false, bsl, (short)0, A, false, false);
                A = __builtin_amdgcn_wmma_f32_16x16x32_bf16(false, sl[r], false, bsh, (short)0, A, false, false);
                A = __builtin_amdgcn_wmma_f32_16x16x32_bf16(false, xh[r], false, bxh, (short)0, A, false, false);
                A = __builtin_amdgcn_wmma_f32_16x16x32_bf16(false, xh[r], false, bxl, (short)0, A, false, false);
                A = __builtin_amdgcn_wmma_f32_16x16x32_bf16(false, xl[r], false, bxh, (short)0, A, false, false);
                acc[r][c] = A;
            }
        }
    }

    // Epilogue: weighted[n,k] = -0.5*quad + ck[k]; logsumexp over 64 cols.
    // C layout: lane L holds column N=L%16; VGPR v holds row +v (L<16) or +8+v (L>=16).
    const float ck0 = ck[col0], ck1 = ck[16 + col0], ck2 = ck[32 + col0], ck3 = ck[48 + col0];
    float wsum = 0.0f;
#pragma unroll
    for (int r = 0; r < 2; ++r) {
        const int rbase = base + r * 16;
#pragma unroll
        for (int v = 0; v < 8; ++v) {
            float t0 = -0.5f * acc[r][0][v] + ck0;
            float t1 = -0.5f * acc[r][1][v] + ck1;
            float t2 = -0.5f * acc[r][2][v] + ck2;
            float t3 = -0.5f * acc[r][3][v] + ck3;
            float mx = fmaxf(fmaxf(t0, t1), fmaxf(t2, t3));
#pragma unroll
            for (int mk = 1; mk < 16; mk <<= 1) mx = fmaxf(mx, __shfl_xor(mx, mk, 32));
            float s = expf(t0 - mx) + expf(t1 - mx) + expf(t2 - mx) + expf(t3 - mx);
#pragma unroll
            for (int mk = 1; mk < 16; mk <<= 1) s += __shfl_xor(s, mk, 32);
            float tot = mx + logf(s);
            int rr = rbase + v + ((lane >= 16) ? 8 : 0);
            if ((lane == 0 || lane == 16) && rr < N) wsum += tot;
        }
    }
#pragma unroll
    for (int mk = 16; mk >= 1; mk >>= 1) wsum += __shfl_xor(wsum, mk, 32);
    if (lane == 0) wsum_arr[wave] = wsum;
    __syncthreads();
    if (tid == 0) {
        float b = 0.0f;
        for (int i = 0; i < 8; ++i) b += wsum_arr[i];
        partials[blockIdx.x] = b;
    }
}

// ---------------------------------------------------------------------------
// Deterministic finalize: mean over N
// ---------------------------------------------------------------------------
__global__ void __launch_bounds__(256)
gmm_finalize(const float* __restrict__ partials, int nblocks,
             float* __restrict__ out, int N) {
    __shared__ float sm[256];
    float s = 0.0f;
    for (int i = threadIdx.x; i < nblocks; i += 256) s += partials[i];
    sm[threadIdx.x] = s;
    __syncthreads();
    for (int off = 128; off > 0; off >>= 1) {
        if ((int)threadIdx.x < off) sm[threadIdx.x] += sm[threadIdx.x + off];
        __syncthreads();
    }
    if (threadIdx.x == 0) out[0] = sm[0] / (float)N;
}

extern "C" void kernel_launch(void* const* d_in, const int* in_sizes, int n_in,
                              void* d_out, int out_size, void* d_ws, size_t ws_size,
                              hipStream_t stream) {
    const float* x  = (const float*)d_in[0];
    const float* w  = (const float*)d_in[1];
    const float* mu = (const float*)d_in[2];
    const float* lc = (const float*)d_in[3];

    const int N = in_sizes[0] / GMM_D;

    // workspace layout (dwords): bhi[16384] | blo[16384] | ck[64] | partials[nblocks]
    uint32_t* bhi      = (uint32_t*)d_ws;
    uint32_t* blo      = bhi + 16384;
    float*    ckbuf    = (float*)(blo + 16384);
    float*    partials = ckbuf + 64;

    const int nblocks = (N + 255) / 256;   // 256 rows per workgroup

    gmm_prep_frags<<<64, 256, 0, stream>>>(mu, lc, bhi, blo);
    gmm_prep_const<<<1, 64, 0, stream>>>(w, mu, lc, ckbuf);
    gmm_main<<<nblocks, 256, 32768 * sizeof(uint32_t), stream>>>(
        x, bhi, blo, ckbuf, partials, N);
    gmm_finalize<<<1, 256, 0, stream>>>(partials, nblocks, (float*)d_out, N);
}
